// PointNet2UNet_68496138437083
// MI455X (gfx1250) — matrix operand order repaired and checked by reading, not compiled, not run on hardware
//
#include <hip/hip_runtime.h>

typedef __attribute__((ext_vector_type(16))) _Float16 v16h;
typedef __attribute__((ext_vector_type(8)))  float    v8f;

#define BATCH 4
#define NS 32            // nsample per ball (== wave32, one lane per sample)

// ---------------------------------------------------------------------------
// Furthest point sampling: one workgroup (1024 thr) per batch. Points and
// running min-distances live in registers (<=16 per thread). Greedy argmax
// via wave shfl reduce + cross-wave LDS reduce. Writes sampled coords.
// ---------------------------------------------------------------------------
__global__ __launch_bounds__(1024)
void fps_kernel(const float* __restrict__ xyz, int N, int S,
                float* __restrict__ new_xyz) {
  const int b = blockIdx.x;
  const float* p = xyz + (size_t)b * N * 3;
  float* nx = new_xyz + (size_t)b * S * 3;
  const int tid = threadIdx.x;
  const int T = 1024;

  float px[16], py[16], pz[16], mind[16];
  int np = 0;
#pragma unroll
  for (int i = 0; i < 16; ++i) {
    int idx = tid + i * T;
    if (idx < N) {
      px[i] = p[idx * 3 + 0]; py[i] = p[idx * 3 + 1]; pz[i] = p[idx * 3 + 2];
      mind[i] = 1e10f; np = i + 1;
    }
  }
  __shared__ float s_sel[3];
  __shared__ float s_wd[32];
  __shared__ int   s_wi[32];
  if (tid == 0) {
    s_sel[0] = p[0]; s_sel[1] = p[1]; s_sel[2] = p[2];
    nx[0] = p[0]; nx[1] = p[1]; nx[2] = p[2];        // first index = 0
  }
  __syncthreads();

  for (int it = 1; it < S; ++it) {
    float cx = s_sel[0], cy = s_sel[1], cz = s_sel[2];
    float bd = -1.0f; int bi = 0x7fffffff;
#pragma unroll
    for (int i = 0; i < 16; ++i) {
      if (i < np) {
        float dx = px[i] - cx, dy = py[i] - cy, dz = pz[i] - cz;
        float d = dx * dx + dy * dy + dz * dz;
        float m = mind[i] < d ? mind[i] : d;
        mind[i] = m;
        int gi = tid + i * T;
        if (m > bd || (m == bd && gi < bi)) { bd = m; bi = gi; }
      }
    }
    // wave32 argmax (max dist, tie -> lowest index, matches jnp.argmax)
    for (int off = 16; off; off >>= 1) {
      float od = __shfl_down(bd, off);
      int   oi = __shfl_down(bi, off);
      if (od > bd || (od == bd && oi < bi)) { bd = od; bi = oi; }
    }
    int w = tid >> 5;
    if ((tid & 31) == 0) { s_wd[w] = bd; s_wi[w] = bi; }
    __syncthreads();
    if (tid < 32) {
      bd = s_wd[tid]; bi = s_wi[tid];
      for (int off = 16; off; off >>= 1) {
        float od = __shfl_down(bd, off);
        int   oi = __shfl_down(bi, off);
        if (od > bd || (od == bd && oi < bi)) { bd = od; bi = oi; }
      }
      if (tid == 0) {
        float sx = p[bi * 3 + 0], sy = p[bi * 3 + 1], sz = p[bi * 3 + 2];
        s_sel[0] = sx; s_sel[1] = sy; s_sel[2] = sz;
        nx[it * 3 + 0] = sx; nx[it * 3 + 1] = sy; nx[it * 3 + 2] = sz;
      }
    }
    __syncthreads();
  }
}

// ---------------------------------------------------------------------------
// Ball query: one wave per center. Ordered compaction of the first NS hits
// using ballot + prefix popcount (nsample == 32 == wave size). Pad with the
// first hit (each center is itself a point, so >=1 hit is guaranteed).
// ---------------------------------------------------------------------------
__global__ void ball_query_kernel(const float* __restrict__ xyz,
                                  const float* __restrict__ nxyz,
                                  int N, int S, int centers, float r2,
                                  int* __restrict__ out) {
  int gw   = (blockIdx.x * blockDim.x + threadIdx.x) >> 5;
  int lane = threadIdx.x & 31;
  if (gw >= centers) return;
  int b = gw / S;
  const float* p = xyz + (size_t)b * N * 3;
  float cx = nxyz[gw * 3 + 0], cy = nxyz[gw * 3 + 1], cz = nxyz[gw * 3 + 2];
  int* o = out + (size_t)gw * NS;

  int cnt = 0, firstIdx = 0;
  bool haveFirst = false;
  for (int base = 0; base < N && cnt < NS; base += 32) {
    int idx = base + lane;
    bool valid = false;
    if (idx < N) {
      float dx = p[idx * 3 + 0] - cx, dy = p[idx * 3 + 1] - cy, dz = p[idx * 3 + 2] - cz;
      valid = (dx * dx + dy * dy + dz * dz) < r2;
    }
    unsigned long long m = __ballot(valid);
    if (!haveFirst && m) { firstIdx = base + (__ffsll((unsigned long long)m) - 1); haveFirst = true; }
    if (valid) {
      int pos = cnt + __popcll(m & ((1ull << lane) - 1ull));
      if (pos < NS) o[pos] = idx;
    }
    cnt += __popcll(m);
  }
  if (cnt < NS && lane >= cnt) o[lane] = firstIdx;
}

// ---------------------------------------------------------------------------
// SA gather: build column-major f16 activation matrix X[col*Kpad + k],
// col=(b,s,n): rows 0..2 = relative xyz, 3..3+C-1 = gathered features, pad 0.
// Column-major K makes WMMA B-fragment loads contiguous per lane.
// ---------------------------------------------------------------------------
__global__ void sa_gather_kernel(const float* __restrict__ xyz,
                                 const float* __restrict__ nxyz,
                                 const int* __restrict__ bidx,
                                 const float* __restrict__ feats,
                                 int N, int S, int C, int Kpad, int cols,
                                 _Float16* __restrict__ X) {
  int col = blockIdx.x * blockDim.x + threadIdx.x;
  if (col >= cols) return;
  int sc = col >> 5;            // (b,s)
  int s = sc % S, b = sc / S;
  int pidx = bidx[col];
  const float* P = xyz + ((size_t)b * N + pidx) * 3;
  const float* Q = nxyz + ((size_t)sc) * 3;
  _Float16* xc = X + (size_t)col * Kpad;
  xc[0] = (_Float16)(P[0] - Q[0]);
  xc[1] = (_Float16)(P[1] - Q[1]);
  xc[2] = (_Float16)(P[2] - Q[2]);
  for (int c = 0; c < C; ++c)
    xc[3 + c] = (_Float16)feats[((size_t)b * C + c) * N + pidx];
  for (int k = 3 + C; k < Kpad; ++k) xc[k] = (_Float16)0.0f;
}

// Weight f32 -> f16, K padded to multiple of 32 with zeros.
__global__ void convert_w_kernel(const float* __restrict__ W, int O, int K, int Kpad,
                                 _Float16* __restrict__ Wf) {
  int i = blockIdx.x * blockDim.x + threadIdx.x;
  if (i >= O * Kpad) return;
  int o = i / Kpad, k = i % Kpad;
  Wf[i] = (_Float16)(k < K ? W[o * K + k] : 0.0f);
}

// ---------------------------------------------------------------------------
// WMMA GEMM: H(O x cols) = Wf(O x Kpad) * X(Kpad x cols, column-major) + bias.
// One wave per 16x16 output tile, K stepped by 32 via v_wmma_f32_16x16x32_f16.
// Fragment layouts used here (contiguous k for A and B, not the ISA table order):
//   A 16x32 f16 : lanes 0-15 M=lane, K=kk..kk+15;
//                 lanes 16-31 same M, K=kk+16..kk+31.
//   B 32x16 f16 : lanes 0-15 N=lane, K=kk..kk+15; lanes 16-31 K=kk+16..kk+31.
//   D 16x16 f32 : vgpr r -> M = r (+8 for upper half-wave), N = lane&15.
// ---------------------------------------------------------------------------
__global__ void wmma_gemm_kernel(const _Float16* __restrict__ Wf,
                                 const _Float16* __restrict__ X,
                                 const float* __restrict__ bias,
                                 int O, int Kpad, int cols,
                                 float* __restrict__ H) {
  int wavesPerBlock = blockDim.x >> 5;
  int wid  = blockIdx.x * wavesPerBlock + (threadIdx.x >> 5);
  int lane = threadIdx.x & 31;
  int Mt = O >> 4, Nt = cols >> 4;
  if (wid >= Mt * Nt) return;
  int mt = wid % Mt, nt = wid / Mt;
  int obase = mt << 4, cbase = nt << 4;
  int lm = lane & 15, lh = lane >> 4;

  const _Float16* wrow = Wf + (size_t)(obase + lm) * Kpad;
  const _Float16* xcol = X + (size_t)(cbase + lm) * Kpad;

  v8f acc = {};
  for (int kk = 0; kk < Kpad; kk += 32) {
    v16h a, bf;
#pragma unroll
    for (int i = 0; i < 8; ++i) a[i] = wrow[kk + lh * 16 + i];
#pragma unroll
    for (int i = 0; i < 8; ++i) a[8 + i] = wrow[kk + lh * 16 + 8 + i];
#pragma unroll
    for (int i = 0; i < 16; ++i) bf[i] = xcol[kk + lh * 16 + i];
    acc = __builtin_amdgcn_wmma_f32_16x16x32_f16(
        /*neg_a=*/false, a, /*neg_b=*/false, bf,
        /*c_mod=*/(short)0, acc, /*reuse_a=*/false, /*reuse_b=*/false);
  }
  int col = cbase + lm;
#pragma unroll
  for (int r = 0; r < 8; ++r) {
    int o = obase + r + lh * 8;
    H[(size_t)o * cols + col] = acc[r] + bias[o];
  }
}

// ---------------------------------------------------------------------------
// Max over the 32 samples of each (o, b, s) group + fp32 sum/sumsq of the
// PRE-pool values per channel (BN stats are over all B*S*NS elements).
// gamma>0 so BN+ReLU commute with max -> pool first, normalize after.
// ---------------------------------------------------------------------------
__global__ void sa_max_reduce_kernel(const float* __restrict__ H, int cols2,
                                     float* __restrict__ Hmax,
                                     float* __restrict__ sums) {
  int o = blockIdx.y;
  int j = blockIdx.x * blockDim.x + threadIdx.x;
  float a = 0.f, b2 = 0.f;
  if (j < cols2) {
    const float* h = H + ((size_t)o * cols2 + j) * NS;
    float m = -1e30f;
#pragma unroll
    for (int n = 0; n < NS; ++n) {
      float v = h[n];
      m = v > m ? v : m;
      a += v; b2 += v * v;
    }
    Hmax[(size_t)o * cols2 + j] = m;
  }
  for (int off = 16; off; off >>= 1) { a += __shfl_down(a, off); b2 += __shfl_down(b2, off); }
  __shared__ float s0[8], s1[8];
  int w = threadIdx.x >> 5;
  if ((threadIdx.x & 31) == 0) { s0[w] = a; s1[w] = b2; }
  __syncthreads();
  if (threadIdx.x == 0) {
    float t0 = 0.f, t1 = 0.f;
    for (int i = 0; i < 8; ++i) { t0 += s0[i]; t1 += s1[i]; }
    atomicAdd(&sums[2 * o + 0], t0);
    atomicAdd(&sums[2 * o + 1], t1);
  }
}

// Per-channel sum/sumsq of an (O x cols) matrix (BN stats for FP / final).
__global__ void col_sums_kernel(const float* __restrict__ A, int cols,
                                float* __restrict__ sums) {
  int o = blockIdx.y;
  int j = blockIdx.x * blockDim.x + threadIdx.x;
  float v = (j < cols) ? A[(size_t)o * cols + j] : 0.f;
  float a = v, b2 = v * v;
  for (int off = 16; off; off >>= 1) { a += __shfl_down(a, off); b2 += __shfl_down(b2, off); }
  __shared__ float s0[8], s1[8];
  int w = threadIdx.x >> 5;
  if ((threadIdx.x & 31) == 0) { s0[w] = a; s1[w] = b2; }
  __syncthreads();
  if (threadIdx.x == 0) {
    float t0 = 0.f, t1 = 0.f;
    for (int i = 0; i < 8; ++i) { t0 += s0[i]; t1 += s1[i]; }
    atomicAdd(&sums[2 * o + 0], t0);
    atomicAdd(&sums[2 * o + 1], t1);
  }
}

// BN (train mode, population stats from sums) + ReLU.
// outMode 0 : write (B, O, Ncur) feature layout   (feeds next stage / d_out)
// outMode 1 : keep (O, B*Ncur) matrix layout      (feeds another col_sums)
__global__ void bn_relu_H_kernel(const float* __restrict__ A,
                                 const float* __restrict__ sums, float count,
                                 const float* __restrict__ gamma,
                                 const float* __restrict__ beta,
                                 int O, int Ncur, int outMode,
                                 float* __restrict__ out) {
  int o = blockIdx.y;
  int j = blockIdx.x * blockDim.x + threadIdx.x;
  int cols = BATCH * Ncur;
  if (j >= cols) return;
  float mu  = sums[2 * o + 0] / count;
  float var = sums[2 * o + 1] / count - mu * mu;
  float sc  = gamma[o] * rsqrtf(var + 1e-5f);
  float v = (A[(size_t)o * cols + j] - mu) * sc + beta[o];
  v = v > 0.f ? v : 0.f;
  if (outMode == 0) {
    int b = j / Ncur, i = j % Ncur;
    out[((size_t)b * O + o) * Ncur + i] = v;
  } else {
    out[(size_t)o * cols + j] = v;
  }
}

// ---------------------------------------------------------------------------
// FP: brute-force 3-NN (running top-3, ties -> earlier index like top_k),
// inverse-sq-distance interpolation of coarse features, concat fine features,
// emit column-major f16 X for the WMMA GEMM.
// ---------------------------------------------------------------------------
__global__ void three_interp_kernel(const float* __restrict__ xyzF,
                                    const float* __restrict__ xyzC,
                                    const float* __restrict__ featC,
                                    const float* __restrict__ featF,
                                    int N1, int N2, int C2, int C1, int Kpad,
                                    _Float16* __restrict__ X) {
  int g = blockIdx.x * blockDim.x + threadIdx.x;   // col = b*N1 + i
  if (g >= BATCH * N1) return;
  int b = g / N1, i = g % N1;
  float qx = xyzF[(size_t)g * 3 + 0], qy = xyzF[(size_t)g * 3 + 1], qz = xyzF[(size_t)g * 3 + 2];
  const float* p2 = xyzC + (size_t)b * N2 * 3;
  float d0 = 1e30f, d1 = 1e30f, d2 = 1e30f;
  int i0 = 0, i1 = 0, i2 = 0;
  for (int k = 0; k < N2; ++k) {
    float dx = p2[k * 3 + 0] - qx, dy = p2[k * 3 + 1] - qy, dz = p2[k * 3 + 2] - qz;
    float d = dx * dx + dy * dy + dz * dz;
    if (d < d0)      { d2 = d1; i2 = i1; d1 = d0; i1 = i0; d0 = d; i0 = k; }
    else if (d < d1) { d2 = d1; i2 = i1; d1 = d;  i1 = k; }
    else if (d < d2) { d2 = d;  i2 = k; }
  }
  float w0 = 1.f / (d0 + 1e-8f), w1 = 1.f / (d1 + 1e-8f), w2 = 1.f / (d2 + 1e-8f);
  float ws = w0 + w1 + w2;
  w0 /= ws; w1 /= ws; w2 /= ws;

  _Float16* xc = X + (size_t)g * Kpad;
  const float* f2 = featC + (size_t)b * C2 * N2;
  for (int c = 0; c < C2; ++c) {
    float v = f2[(size_t)c * N2 + i0] * w0 + f2[(size_t)c * N2 + i1] * w1 +
              f2[(size_t)c * N2 + i2] * w2;
    xc[c] = (_Float16)v;
  }
  const float* f1 = featF + (size_t)b * C1 * N1;
  for (int c = 0; c < C1; ++c) xc[C2 + c] = (_Float16)f1[(size_t)c * N1 + i];
  for (int k = C2 + C1; k < Kpad; ++k) xc[k] = (_Float16)0.0f;
}

// ---------------------------------------------------------------------------
// Host driver
// ---------------------------------------------------------------------------
extern "C" void kernel_launch(void* const* d_in, const int* in_sizes, int n_in,
                              void* d_out, int out_size, void* d_ws, size_t ws_size,
                              hipStream_t stream) {
  (void)in_sizes; (void)n_in; (void)out_size; (void)ws_size;

  const float* xyz0  = (const float*)d_in[0];
  const float* feat0 = (const float*)d_in[1];
  auto saP = [&](int s, int k) { return (const float*)d_in[2 + s * 4 + k]; };
  auto fpP = [&](int s, int k) { return (const float*)d_in[18 + s * 4 + k]; };
  const float* outG = (const float*)d_in[34];
  const float* outB = (const float*)d_in[35];

  char* ws = (char*)d_ws;
  size_t off = 0;
  auto alloc = [&](size_t bytes) {
    void* p = ws + off;
    off += (bytes + 255) & ~(size_t)255;
    return p;
  };
  _Float16* Xf   = (_Float16*)alloc((size_t)8 << 20);   // max 32*131072 / 64*65536 f16
  float*    H    = (float*)   alloc((size_t)16 << 20);  // max 32*131072 f32
  float*    Hmax = (float*)   alloc((size_t)1 << 20);
  _Float16* Wf   = (_Float16*)alloc((size_t)1 << 20);
  float*    sums = (float*)   alloc(4096);
  int*      bidx = (int*)     alloc((size_t)1 << 20);
  float* nx1 = (float*)alloc((size_t)BATCH * 1024 * 3 * 4);
  float* nx2 = (float*)alloc((size_t)BATCH * 256 * 3 * 4);
  float* nx3 = (float*)alloc((size_t)BATCH * 64 * 3 * 4);
  float* nx4 = (float*)alloc((size_t)BATCH * 16 * 3 * 4);
  float* f1  = (float*)alloc((size_t)BATCH * 32 * 1024 * 4);
  float* f2  = (float*)alloc((size_t)BATCH * 64 * 256 * 4);
  float* f3  = (float*)alloc((size_t)BATCH * 128 * 64 * 4);
  float* f4  = (float*)alloc((size_t)BATCH * 256 * 16 * 4);
  float* g3  = (float*)alloc((size_t)BATCH * 128 * 64 * 4);
  float* g2  = (float*)alloc((size_t)BATCH * 64 * 256 * 4);
  float* g1  = (float*)alloc((size_t)BATCH * 32 * 1024 * 4);
  float* g0  = (float*)alloc((size_t)32 * BATCH * 16384 * 4);  // (O, B*N) layout

  auto run_sa = [&](const float* pts, const float* fin, int Nin, int Cin,
                    int S, float r2, int O, int K, int Kpad,
                    const float* W, const float* bias, const float* gma, const float* bta,
                    float* nxyz, float* fout) {
    fps_kernel<<<BATCH, 1024, 0, stream>>>(pts, Nin, S, nxyz);
    int centers = BATCH * S;
    ball_query_kernel<<<(centers * 32 + 255) / 256, 256, 0, stream>>>(
        pts, nxyz, Nin, S, centers, r2, bidx);
    int cols = centers * NS;
    sa_gather_kernel<<<(cols + 255) / 256, 256, 0, stream>>>(
        pts, nxyz, bidx, fin, Nin, S, Cin, Kpad, cols, Xf);
    convert_w_kernel<<<(O * Kpad + 255) / 256, 256, 0, stream>>>(W, O, K, Kpad, Wf);
    int tiles = (O / 16) * (cols / 16);
    wmma_gemm_kernel<<<(tiles + 7) / 8, 256, 0, stream>>>(Wf, Xf, bias, O, Kpad, cols, H);
    (void)hipMemsetAsync(sums, 0, 2 * O * sizeof(float), stream);
    int cols2 = centers;
    dim3 gr((cols2 + 255) / 256, O);
    sa_max_reduce_kernel<<<gr, 256, 0, stream>>>(H, cols2, Hmax, sums);
    bn_relu_H_kernel<<<gr, 256, 0, stream>>>(Hmax, sums, (float)cols2 * NS,
                                             gma, bta, O, S, 0, fout);
  };

  auto run_fp = [&](const float* xyzF, const float* xyzC, const float* featC,
                    const float* featF, int N1, int N2, int C2, int C1,
                    int O, int K, int Kpad,
                    const float* W, const float* bias, const float* gma, const float* bta,
                    int outMode, float* fout) {
    int cols = BATCH * N1;
    three_interp_kernel<<<(cols + 255) / 256, 256, 0, stream>>>(
        xyzF, xyzC, featC, featF, N1, N2, C2, C1, Kpad, Xf);
    convert_w_kernel<<<(O * Kpad + 255) / 256, 256, 0, stream>>>(W, O, K, Kpad, Wf);
    int tiles = (O / 16) * (cols / 16);
    wmma_gemm_kernel<<<(tiles + 7) / 8, 256, 0, stream>>>(Wf, Xf, bias, O, Kpad, cols, H);
    (void)hipMemsetAsync(sums, 0, 2 * O * sizeof(float), stream);
    dim3 gr((cols + 255) / 256, O);
    col_sums_kernel<<<gr, 256, 0, stream>>>(H, cols, sums);
    bn_relu_H_kernel<<<gr, 256, 0, stream>>>(H, sums, (float)cols, gma, bta,
                                             O, N1, outMode, fout);
  };

  // --- Set abstraction (Kpad = roundup32(K)) ---
  run_sa(xyz0, feat0, 16384,   6, 1024, 0.01f,  32,   9,  32,
         saP(0,0), saP(0,1), saP(0,2), saP(0,3), nx1, f1);
  run_sa(nx1,  f1,    1024,  32,  256, 0.04f,  64,  35,  64,
         saP(1,0), saP(1,1), saP(1,2), saP(1,3), nx2, f2);
  run_sa(nx2,  f2,     256,  64,   64, 0.16f, 128,  67,  96,
         saP(2,0), saP(2,1), saP(2,2), saP(2,3), nx3, f3);
  run_sa(nx3,  f3,      64, 128,   16, 0.64f, 256, 131, 160,
         saP(3,0), saP(3,1), saP(3,2), saP(3,3), nx4, f4);

  // --- Feature propagation (coarse -> fine) ---
  run_fp(nx3, nx4, f4, f3,      64,   16, 256, 128, 128, 384, 384,
         fpP(0,0), fpP(0,1), fpP(0,2), fpP(0,3), 0, g3);
  run_fp(nx2, nx3, g3, f2,     256,   64, 128,  64,  64, 192, 192,
         fpP(1,0), fpP(1,1), fpP(1,2), fpP(1,3), 0, g2);
  run_fp(nx1, nx2, g2, f1,    1024,  256,  64,  32,  32,  96,  96,
         fpP(2,0), fpP(2,1), fpP(2,2), fpP(2,3), 0, g1);
  run_fp(xyz0, nx1, g1, feat0, 16384, 1024, 32,  6,  32,  38,  64,
         fpP(3,0), fpP(3,1), fpP(3,2), fpP(3,3), 1, g0);  // keep (O, B*N) layout

  // --- Final BN + ReLU -> d_out in (B, 32, 16384) ---
  (void)hipMemsetAsync(sums, 0, 2 * 32 * sizeof(float), stream);
  dim3 gf((BATCH * 16384 + 255) / 256, 32);
  col_sums_kernel<<<gf, 256, 0, stream>>>(g0, BATCH * 16384, sums);
  bn_relu_H_kernel<<<gf, 256, 0, stream>>>(g0, sums, (float)(BATCH * 16384),
                                           outG, outB, 32, 16384, 0, (float*)d_out);
}
